// GraphMSG_52699248722389
// MI455X (gfx1250) — matrix operand order repaired and checked by reading, not compiled, not run on hardware
//
#include <hip/hip_runtime.h>
#include <cstddef>

// ---------------- constants from the reference ----------------
#define ERA_N   40320
#define H_N     10242
#define C_N     256
#define F_N     99
#define OUT_W   80
#define NLAYERS 6
#define E1_N    122880   // e2h
#define E2_N    120960   // h2e
#define E3_N    61440    // h2h

typedef __attribute__((ext_vector_type(16))) __bf16 v16bf;
typedef __attribute__((ext_vector_type(8)))  float  v8f;

__device__ __forceinline__ unsigned short f2b(float f) {
  union { float f; unsigned u; } v; v.f = f;
  unsigned u = v.u;
  u += 0x7FFFu + ((u >> 16) & 1u);          // round-to-nearest-even
  return (unsigned short)(u >> 16);
}

// ---------------------------------------------------------------------------
// Generic bf16 WMMA GEMM:  Y[M,Npad] = act(A[M,Ktot] @ W[Ktot,Npad] + bias)
// A is a logical concat of up to 3 K-segments (each a multiple of 32), with
// optional per-row gather indices for segments 0/1 (segment 2 is direct).
// W is pre-transposed+padded to Wt[Npad, Ktot] bf16 in natural k order
// (see prep_wt_kernel). Npad is a multiple of 64,
// so ALL four 16-col tiles per wave are unconditionally live: the K loop has
// no branches, keeping accumulators pinned and EXEC all-ones for WMMA.
// Stores (and bias reads) are guarded by Nstore (the real output width).
// Block = 128 threads (4 waves); wave w -> rows [m0w..m0w+15], cols [n0..n0+63].
// ---------------------------------------------------------------------------
__global__ __launch_bounds__(128)
void gemm_bf16_kernel(
    const unsigned short* __restrict__ A0, const int* __restrict__ I0, int K0, int lda0,
    const unsigned short* __restrict__ A1, const int* __restrict__ I1, int K1, int lda1,
    const unsigned short* __restrict__ A2, int K2, int lda2,
    const unsigned short* __restrict__ Wt, int Ktot,
    const float* __restrict__ bias,
    float* __restrict__ Yf, int ldyf,
    unsigned short* __restrict__ Yb, int ldyb,
    int M, int Nstore, int act)
{
  const int lane = threadIdx.x & 31;
  const int wave = threadIdx.x >> 5;
  const int m0   = blockIdx.x * 64 + wave * 16;
  const int n0   = blockIdx.y * 64;
  const int hi   = lane >> 4;      // 0: lanes 0-15 (K lo half), 1: lanes 16-31 (K hi half)
  const int lr   = lane & 15;

  int gm = m0 + lr; if (gm > M - 1) gm = M - 1;   // arithmetic clamp (no EXEC divergence)

  const int r0 = I0 ? I0[gm] : gm;
  const unsigned short* arow0 = A0 + (size_t)r0 * lda0 + hi * 16;
  const unsigned short* arow1 = arow0;
  if (K1) { int r1 = I1 ? I1[gm] : gm; arow1 = A1 + (size_t)r1 * lda1 + hi * 16; }
  const unsigned short* arow2 = arow0;
  if (K2) { arow2 = A2 + (size_t)gm * lda2 + hi * 16; }

  const unsigned short* wcol = Wt + (size_t)(n0 + lr) * Ktot + hi * 16;
  const size_t wts = (size_t)16 * Ktot;          // Wt stride for one 16-col tile

  v8f acc0 = {}, acc1 = {}, acc2 = {}, acc3 = {};

#define KSTEP(AP, KW)                                                          \
  do {                                                                         \
    v16bf av = *(const v16bf*)(AP);                                            \
    v16bf b0 = *(const v16bf*)(wcol + (KW));                                   \
    v16bf b1 = *(const v16bf*)(wcol + wts + (KW));                             \
    v16bf b2 = *(const v16bf*)(wcol + 2 * wts + (KW));                         \
    v16bf b3 = *(const v16bf*)(wcol + 3 * wts + (KW));                         \
    acc0 = __builtin_amdgcn_wmma_f32_16x16x32_bf16(false, av, false, b0,       \
                                                   (short)0, acc0, false, false); \
    acc1 = __builtin_amdgcn_wmma_f32_16x16x32_bf16(false, av, false, b1,       \
                                                   (short)0, acc1, false, false); \
    acc2 = __builtin_amdgcn_wmma_f32_16x16x32_bf16(false, av, false, b2,       \
                                                   (short)0, acc2, false, false); \
    acc3 = __builtin_amdgcn_wmma_f32_16x16x32_bf16(false, av, false, b3,       \
                                                   (short)0, acc3, false, false); \
  } while (0)

  for (int kb = 0; kb < K0; kb += 32) KSTEP(arow0 + kb, kb);
  for (int kb = 0; kb < K1; kb += 32) KSTEP(arow1 + kb, K0 + kb);
  for (int kb = 0; kb < K2; kb += 32) KSTEP(arow2 + kb, K0 + K1 + kb);
#undef KSTEP

  // ----- epilogue: bias + optional SiLU + guarded stores -----
  const v8f accs[4] = {acc0, acc1, acc2, acc3};
#pragma unroll
  for (int nt = 0; nt < 4; ++nt) {
    const int gn = n0 + nt * 16 + lr;
    const bool ncol_ok = gn < Nstore;
    const float bv = (bias && ncol_ok) ? bias[gn] : 0.f;
#pragma unroll
    for (int r = 0; r < 8; ++r) {
      const int gmr = m0 + r + hi * 8;           // C/D layout: VGPR r -> M=r (+8 hi group)
      float y = accs[nt][r] + bv;
      if (act == 1) y = y * (1.f / (1.f + __expf(-y)));   // SiLU
      if (gmr < M && ncol_ok) {
        if (Yf) Yf[(size_t)gmr * ldyf + gn] = y;
        if (Yb) Yb[(size_t)gmr * ldyb + gn] = f2b(y);
      }
    }
  }
}

// ---------------------------------------------------------------------------
// Weight prep: W[K,Nreal] f32 -> Wt[Npad,Kp] bf16 (zero pad K..Kp and
// Nreal..Npad), with k kept in natural order in each 32-block (no slot
// permutation), so contiguous A/B operand loads (k = 16h + s)
// pair matching K elements in the MAC.
// ---------------------------------------------------------------------------
__global__ void prep_wt_kernel(const float* __restrict__ W, int K, int Nreal,
                               unsigned short* __restrict__ Wt, int Kp, int Npad)
{
  size_t i = (size_t)blockIdx.x * blockDim.x + threadIdx.x;
  if (i >= (size_t)Npad * Kp) return;
  int n = (int)(i / Kp);
  int k = (int)(i % Kp);
  int j = k & 31, kb = k & ~31;
  int t = (j >> 3) & 3;
  int sj = (t == 1 || t == 2) ? j : j;    // identity (contiguous k order)
  int ks = kb + sj;
  float v = (ks < K && n < Nreal) ? W[(size_t)ks * Nreal + n] : 0.f;
  Wt[i] = f2b(v);
}

// ---------------- input builders (f32 -> padded bf16 matrices) -------------
__global__ void build_era_in_kernel(const float* __restrict__ x,
                                    const float* __restrict__ ll,
                                    const float* __restrict__ tr,
                                    unsigned short* __restrict__ out)
{
  size_t i = (size_t)blockIdx.x * blockDim.x + threadIdx.x;
  if (i >= (size_t)ERA_N * 224) return;
  int e = (int)(i / 224), c = (int)(i % 224);
  float v = 0.f;
  if      (c < 99)  v = x[(size_t)e * F_N + c];                     // m=0
  else if (c < 198) v = x[(size_t)(ERA_N + e) * F_N + (c - 99)];    // m=1
  else if (c < 202) v = ll[(size_t)e * 4 + (c - 198)];
  else if (c < 210) v = tr[(size_t)e * 8 + (c - 202)];
  out[i] = f2b(v);
}

__global__ void build_cat2_kernel(const float* __restrict__ a, int wa,
                                  const float* __restrict__ b, int wb,
                                  unsigned short* __restrict__ out, int Wp, int M)
{
  size_t i = (size_t)blockIdx.x * blockDim.x + threadIdx.x;
  if (i >= (size_t)M * Wp) return;
  int e = (int)(i / Wp), c = (int)(i % Wp);
  float v = 0.f;
  if      (c < wa)      v = a[(size_t)e * wa + c];
  else if (c < wa + wb) v = b[(size_t)e * wb + (c - wa)];
  out[i] = f2b(v);
}

// ---------------- LayerNorm (in-place f32, optional bf16 copy) -------------
__global__ __launch_bounds__(256)
void layernorm_kernel(float* __restrict__ X, const float* __restrict__ g,
                      const float* __restrict__ be,
                      unsigned short* __restrict__ Xb, int ldb, int M)
{
  __shared__ float sm[256];
  int row = blockIdx.x, t = threadIdx.x;
  (void)M;
  float v = X[(size_t)row * C_N + t];
  sm[t] = v; __syncthreads();
  for (int s = 128; s > 0; s >>= 1) { if (t < s) sm[t] += sm[t + s]; __syncthreads(); }
  float mean = sm[0] * (1.f / C_N); __syncthreads();
  float d = v - mean;
  sm[t] = d * d; __syncthreads();
  for (int s = 128; s > 0; s >>= 1) { if (t < s) sm[t] += sm[t + s]; __syncthreads(); }
  float var = sm[0] * (1.f / C_N);
  float y = d * rsqrtf(var + 1e-5f) * g[t] + be[t];
  X[(size_t)row * C_N + t] = y;
  if (Xb) Xb[(size_t)row * ldb + t] = f2b(y);
}

// ---------------- elementwise helpers --------------------------------------
__global__ void zero_kernel(float* __restrict__ p, size_t n)
{ size_t i = (size_t)blockIdx.x * blockDim.x + threadIdx.x; if (i < n) p[i] = 0.f; }

__global__ void scatter_add_kernel(const float* __restrict__ m, const int* __restrict__ dst,
                                   float* __restrict__ agg)
{
  int e = blockIdx.x, c = threadIdx.x;
  atomicAdd(&agg[(size_t)dst[e] * C_N + c], m[(size_t)e * C_N + c]);
}

__global__ void tob_kernel(const float* __restrict__ a, unsigned short* __restrict__ b, size_t n)
{ size_t i = (size_t)blockIdx.x * blockDim.x + threadIdx.x; if (i < n) b[i] = f2b(a[i]); }

__global__ void addbf_kernel(float* __restrict__ a, const float* __restrict__ b,
                             unsigned short* __restrict__ abf, size_t n)
{
  size_t i = (size_t)blockIdx.x * blockDim.x + threadIdx.x;
  if (i >= n) return;
  float v = a[i] + b[i];
  a[i] = v;
  if (abf) abf[i] = f2b(v);
}

__global__ void resid2_kernel(const float* __restrict__ base, const float* __restrict__ delta,
                              float* __restrict__ o1f, unsigned short* __restrict__ o1b,
                              float* __restrict__ o2f, unsigned short* __restrict__ o2b, size_t n)
{
  size_t i = (size_t)blockIdx.x * blockDim.x + threadIdx.x;
  if (i >= n) return;
  float v = base[i] + delta[i];
  if (o1f) o1f[i] = v;
  if (o1b) o1b[i] = f2b(v);
  if (o2f) o2f[i] = v;
  if (o2b) o2b[i] = f2b(v);
}

// ===========================================================================
extern "C" void kernel_launch(void* const* d_in, const int* in_sizes, int n_in,
                              void* d_out, int out_size, void* d_ws, size_t ws_size,
                              hipStream_t stream)
{
  (void)in_sizes; (void)out_size;
  if (n_in < 159 || !d_ws) return;

  const float* x_in     = (const float*)d_in[0];
  const float* era_ll   = (const float*)d_in[1];
  const float* h_ll     = (const float*)d_in[2];
  const float* era_tr   = (const float*)d_in[3];
  const float* h_tr     = (const float*)d_in[4];
  const float* e2h_tr   = (const float*)d_in[5];
  const float* h2e_tr   = (const float*)d_in[6];
  const float* h2h_tr   = (const float*)d_in[7];
  const float* e2h_attr = (const float*)d_in[8];
  const float* h2e_attr = (const float*)d_in[9];
  const float* h2h_attr = (const float*)d_in[10];
  const int* e2h_src = (const int*)d_in[11];
  const int* e2h_dst = (const int*)d_in[12];
  const int* h2e_src = (const int*)d_in[13];
  const int* h2e_dst = (const int*)d_in[14];
  const int* h2h_src = (const int*)d_in[15];
  const int* h2h_dst = (const int*)d_in[16];

  // --- params: jax pytree flatten = alphabetical dict keys, lists in order.
  // Top: bwd < fwd < proc. Per-MLP keys sort as: W[2], b[2], be, g.
  struct M2 { const float *W0, *W1, *b0, *b1, *be, *g; };
  int pi = 17;
  auto nf = [&]() { return (const float*)d_in[pi++]; };
  auto getmlp = [&](bool ln) {
    M2 m; m.W0 = nf(); m.W1 = nf(); m.b0 = nf(); m.b1 = nf();
    m.be = ln ? nf() : nullptr; m.g = ln ? nf() : nullptr; return m;
  };
  M2 bwd_dst = getmlp(true), bwd_edge = getmlp(true), bwd_msg = getmlp(true),
     bwd_node = getmlp(true), bwd_outp = getmlp(false), bwd_src = getmlp(true);
  M2 fwd_dst = getmlp(true), fwd_edge = getmlp(true), fwd_msg = getmlp(true),
     fwd_node = getmlp(true), fwd_src = getmlp(true);
  M2 proc_edge = getmlp(true);
  M2 proc_msg[NLAYERS], proc_node[NLAYERS];
  for (int l = 0; l < NLAYERS; ++l) proc_msg[l]  = getmlp(true);
  for (int l = 0; l < NLAYERS; ++l) proc_node[l] = getmlp(true);

  // --- workspace bump allocator ---------------------------------------
  char* base = (char*)d_ws;
  size_t off = 0;
  auto alloc = [&](size_t bytes) -> void* {
    size_t o = (off + 255) & ~(size_t)255; off = o + bytes; return (void*)(base + o);
  };

  struct PM { M2 p; unsigned short *Wt0, *Wt1; int K0, K0p, N1, N1p; };
  auto mkpm = [&](const M2& m, int K0, int K0p, int N1, int N1p) {
    PM pm; pm.p = m; pm.K0 = K0; pm.K0p = K0p; pm.N1 = N1; pm.N1p = N1p;
    pm.Wt0 = (unsigned short*)alloc((size_t)C_N * K0p * 2);
    pm.Wt1 = (unsigned short*)alloc((size_t)N1p * C_N * 2);
    return pm;
  };
  PM pm_fwd_src  = mkpm(fwd_src, 210, 224, C_N, C_N);
  PM pm_fwd_dst  = mkpm(fwd_dst,  12,  32, C_N, C_N);
  PM pm_fwd_edge = mkpm(fwd_edge, 11,  32, C_N, C_N);
  PM pm_fwd_msg  = mkpm(fwd_msg, 768, 768, C_N, C_N);
  PM pm_fwd_node = mkpm(fwd_node, 512, 512, C_N, C_N);
  PM pm_proc_edge = mkpm(proc_edge, 11, 32, C_N, C_N);
  PM pm_proc_msg[NLAYERS], pm_proc_node[NLAYERS];
  for (int l = 0; l < NLAYERS; ++l) pm_proc_msg[l]  = mkpm(proc_msg[l], 768, 768, C_N, C_N);
  for (int l = 0; l < NLAYERS; ++l) pm_proc_node[l] = mkpm(proc_node[l], 512, 512, C_N, C_N);
  PM pm_bwd_src  = mkpm(bwd_src, 256, 256, C_N, C_N);
  PM pm_bwd_dst  = mkpm(bwd_dst, 256, 256, C_N, C_N);
  PM pm_bwd_edge = mkpm(bwd_edge, 11,  32, C_N, C_N);
  PM pm_bwd_msg  = mkpm(bwd_msg, 768, 768, C_N, C_N);
  PM pm_bwd_node = mkpm(bwd_node, 512, 512, C_N, C_N);
  PM pm_bwd_out  = mkpm(bwd_outp, 256, 256, OUT_W, 128);   // head: N=80 padded to 128

  const size_t nERA = (size_t)ERA_N * C_N, nH = (size_t)H_N * C_N;
  unsigned short* era_in_b = (unsigned short*)alloc((size_t)ERA_N * 224 * 2);
  unsigned short* hin_b    = (unsigned short*)alloc((size_t)H_N * 32 * 2);
  unsigned short* a1_b     = (unsigned short*)alloc((size_t)E1_N * 32 * 2);
  unsigned short* a2_b     = (unsigned short*)alloc((size_t)E2_N * 32 * 2);
  unsigned short* a3_b     = (unsigned short*)alloc((size_t)E3_N * 32 * 2);
  float*          xeralat_f = (float*)alloc(nERA * 4);
  unsigned short* xeralat_b = (unsigned short*)alloc(nERA * 2);
  float*          xh_f   = (float*)alloc(nH * 4);
  unsigned short* xh_b   = (unsigned short*)alloc(nH * 2);
  float*          xlat_f = (float*)alloc(nH * 4);
  float*          xp_f   = (float*)alloc(nH * 4);
  unsigned short* xp_b   = (unsigned short*)alloc(nH * 2);
  float*          agg_f  = (float*)alloc(nERA * 4);
  unsigned short* agg_b  = (unsigned short*)alloc(nERA * 2);
  float*          eproc_f = (float*)alloc((size_t)E3_N * C_N * 4);
  unsigned short* eproc_b = (unsigned short*)alloc((size_t)E3_N * C_N * 2);
  unsigned short* ebf    = (unsigned short*)alloc((size_t)E1_N * C_N * 2);
  unsigned short* hid_b  = (unsigned short*)alloc((size_t)E1_N * C_N * 2);
  float*          f_pool = (float*)alloc((size_t)E1_N * C_N * 4);
  unsigned short* xs_b   = (unsigned short*)alloc(nH * 2);
  float*          xd_f   = (float*)alloc(nERA * 4);
  unsigned short* xd_b   = (unsigned short*)alloc(nERA * 2);

  if (off > ws_size) return;   // workspace too small: bail before any write

  // --- launch helpers ---------------------------------------------------
  auto g1 = [](size_t n) { return dim3((unsigned)((n + 255) / 256)); };
  auto gemm = [&](const unsigned short* A0, const int* I0, int K0, int lda0,
                  const unsigned short* A1, const int* I1, int K1, int lda1,
                  const unsigned short* A2, int K2, int lda2,
                  const unsigned short* Wt, int Ktot, const float* bias,
                  float* Yf, int ldyf, unsigned short* Yb, int ldyb,
                  int M, int Npad, int Nstore, int act) {
    dim3 grid((unsigned)((M + 63) / 64), (unsigned)(Npad / 64));
    gemm_bf16_kernel<<<grid, 128, 0, stream>>>(A0, I0, K0, lda0, A1, I1, K1, lda1,
                                               A2, K2, lda2, Wt, Ktot, bias,
                                               Yf, ldyf, Yb, ldyb, M, Nstore, act);
  };
  auto lnorm = [&](float* X, const M2& p, unsigned short* Xb, int M) {
    layernorm_kernel<<<dim3((unsigned)M), 256, 0, stream>>>(X, p.g, p.be, Xb, C_N, M);
  };
  // full 2-layer MLP (GEMM-SiLU -> GEMM -> LayerNorm); outf f32 (in-place LN), outb bf16
  auto mlp = [&](PM& pm,
                 const unsigned short* A0, const int* I0, int K0, int lda0,
                 const unsigned short* A1, const int* I1, int K1, int lda1,
                 const unsigned short* A2, int K2, int lda2,
                 int M, float* outf, unsigned short* outb) {
    gemm(A0, I0, K0, lda0, A1, I1, K1, lda1, A2, K2, lda2,
         pm.Wt0, pm.K0p, pm.p.b0, nullptr, 0, hid_b, C_N, M, C_N, C_N, 1);
    gemm(hid_b, nullptr, C_N, C_N, nullptr, nullptr, 0, 0, nullptr, 0, 0,
         pm.Wt1, C_N, pm.p.b1, outf, C_N, nullptr, 0, M, C_N, C_N, 0);
    lnorm(outf, pm.p, outb, M);
  };
  auto zero = [&](float* p, size_t n) { zero_kernel<<<g1(n), 256, 0, stream>>>(p, n); };
  auto scat = [&](const float* m, const int* dst, float* agg, int E) {
    scatter_add_kernel<<<dim3((unsigned)E), 256, 0, stream>>>(m, dst, agg);
  };
  auto tob  = [&](const float* a, unsigned short* b, size_t n) {
    tob_kernel<<<g1(n), 256, 0, stream>>>(a, b, n);
  };
  auto addbf = [&](float* a, const float* b, unsigned short* abf, size_t n) {
    addbf_kernel<<<g1(n), 256, 0, stream>>>(a, b, abf, n);
  };

  // --- weight prep (transpose + pad to bf16) -------------
  auto prep = [&](PM& pm) {
    prep_wt_kernel<<<g1((size_t)C_N * pm.K0p), 256, 0, stream>>>(
        pm.p.W0, pm.K0, C_N, pm.Wt0, pm.K0p, C_N);
    prep_wt_kernel<<<g1((size_t)pm.N1p * C_N), 256, 0, stream>>>(
        pm.p.W1, C_N, pm.N1, pm.Wt1, C_N, pm.N1p);
  };
  prep(pm_fwd_src); prep(pm_fwd_dst); prep(pm_fwd_edge); prep(pm_fwd_msg); prep(pm_fwd_node);
  prep(pm_proc_edge);
  for (int l = 0; l < NLAYERS; ++l) { prep(pm_proc_msg[l]); prep(pm_proc_node[l]); }
  prep(pm_bwd_src); prep(pm_bwd_dst); prep(pm_bwd_edge); prep(pm_bwd_msg);
  prep(pm_bwd_node); prep(pm_bwd_out);

  // --- build padded bf16 inputs ----------------------------------------
  build_era_in_kernel<<<g1((size_t)ERA_N * 224), 256, 0, stream>>>(x_in, era_ll, era_tr, era_in_b);
  build_cat2_kernel<<<g1((size_t)H_N * 32), 256, 0, stream>>>(h_ll, 4, h_tr, 8, hin_b, 32, H_N);
  build_cat2_kernel<<<g1((size_t)E1_N * 32), 256, 0, stream>>>(e2h_attr, 3, e2h_tr, 8, a1_b, 32, E1_N);
  build_cat2_kernel<<<g1((size_t)E2_N * 32), 256, 0, stream>>>(h2e_attr, 3, h2e_tr, 8, a2_b, 32, E2_N);
  build_cat2_kernel<<<g1((size_t)E3_N * 32), 256, 0, stream>>>(h2h_attr, 3, h2h_tr, 8, a3_b, 32, E3_N);

  // ===== encoder (fwd mapper: era -> hidden) ============================
  mlp(pm_fwd_src, era_in_b, nullptr, 224, 224, nullptr, nullptr, 0, 0, nullptr, 0, 0,
      ERA_N, xeralat_f, xeralat_b);
  mlp(pm_fwd_dst, hin_b, nullptr, 32, 32, nullptr, nullptr, 0, 0, nullptr, 0, 0,
      H_N, xh_f, xh_b);
  mlp(pm_fwd_edge, a1_b, nullptr, 32, 32, nullptr, nullptr, 0, 0, nullptr, 0, 0,
      E1_N, f_pool, ebf);
  mlp(pm_fwd_msg, xeralat_b, e2h_src, 256, 256, xh_b, e2h_dst, 256, 256, ebf, 256, 256,
      E1_N, f_pool, nullptr);
  zero(agg_f, nH);
  scat(f_pool, e2h_dst, agg_f, E1_N);
  tob(agg_f, agg_b, nH);
  mlp(pm_fwd_node, xh_b, nullptr, 256, 256, agg_b, nullptr, 256, 256, nullptr, 0, 0,
      H_N, f_pool, nullptr);
  // x_lat = xh + node_out ; xp = x_lat
  resid2_kernel<<<g1(nH), 256, 0, stream>>>(xh_f, f_pool, xlat_f, nullptr, xp_f, xp_b, nH);

  // ===== processor (6 message-passing layers on hidden graph) ===========
  mlp(pm_proc_edge, a3_b, nullptr, 32, 32, nullptr, nullptr, 0, 0, nullptr, 0, 0,
      E3_N, eproc_f, eproc_b);
  for (int l = 0; l < NLAYERS; ++l) {
    mlp(pm_proc_msg[l], xp_b, h2h_src, 256, 256, xp_b, h2h_dst, 256, 256, eproc_b, 256, 256,
        E3_N, f_pool, nullptr);
    addbf(eproc_f, f_pool, eproc_b, (size_t)E3_N * C_N);   // e = e + m
    zero(agg_f, nH);
    scat(f_pool, h2h_dst, agg_f, E3_N);
    tob(agg_f, agg_b, nH);
    mlp(pm_proc_node[l], xp_b, nullptr, 256, 256, agg_b, nullptr, 256, 256, nullptr, 0, 0,
        H_N, f_pool, nullptr);
    addbf(xp_f, f_pool, xp_b, nH);                         // xp = xp + node_out
  }
  addbf(xp_f, xlat_f, xp_b, nH);                           // skip around processor

  // ===== decoder (bwd mapper: hidden -> era -> OUT channels) ============
  mlp(pm_bwd_src, xp_b, nullptr, 256, 256, nullptr, nullptr, 0, 0, nullptr, 0, 0,
      H_N, f_pool, xs_b);
  mlp(pm_bwd_dst, xeralat_b, nullptr, 256, 256, nullptr, nullptr, 0, 0, nullptr, 0, 0,
      ERA_N, xd_f, xd_b);
  mlp(pm_bwd_edge, a2_b, nullptr, 32, 32, nullptr, nullptr, 0, 0, nullptr, 0, 0,
      E2_N, f_pool, ebf);
  mlp(pm_bwd_msg, xs_b, h2e_src, 256, 256, xd_b, h2e_dst, 256, 256, ebf, 256, 256,
      E2_N, f_pool, nullptr);
  zero(agg_f, nERA);
  scat(f_pool, h2e_dst, agg_f, E2_N);
  tob(agg_f, agg_b, nERA);
  mlp(pm_bwd_node, xd_b, nullptr, 256, 256, agg_b, nullptr, 256, 256, nullptr, 0, 0,
      ERA_N, f_pool, nullptr);
  addbf(xd_f, f_pool, xd_b, nERA);                         // xd = xd + node_out

  // output head: Linear-SiLU-Linear (no LayerNorm), f32 straight to d_out
  gemm(xd_b, nullptr, 256, 256, nullptr, nullptr, 0, 0, nullptr, 0, 0,
       pm_bwd_out.Wt0, 256, bwd_outp.b0, nullptr, 0, hid_b, C_N, ERA_N, C_N, C_N, 1);
  gemm(hid_b, nullptr, 256, 256, nullptr, nullptr, 0, 0, nullptr, 0, 0,
       pm_bwd_out.Wt1, 256, bwd_outp.b1, (float*)d_out, OUT_W, nullptr, 0,
       ERA_N, 128, OUT_W, 0);
}
